// DGCRNN_67929202754127
// MI455X (gfx1250) — matrix operand order repaired and checked by reading, not compiled, not run on hardware
//
#include <hip/hip_runtime.h>
#include <hip/hip_bf16.h>

typedef __attribute__((ext_vector_type(16))) _Float16 v16h;
typedef __attribute__((ext_vector_type(8)))  _Float16 v8h;
typedef __attribute__((ext_vector_type(8)))  float    v8f;

union frag16 { v16h v; v8h h[2]; };

// ---- WMMA fragment helpers (CDNA5 wave32 layouts, CDNA5 ISA) ----

// A matrix 16x32 f16, row-major source, row stride `stride` halfs (16B aligned).
// lane<16 -> M=lane, K={0..7,16..23}; lane>=16 -> M=lane-16, K={8..15,24..31}
// Per lane: two contiguous 8-half runs -> two b128 loads.
__device__ __forceinline__ v16h load_a_frag(const _Float16* base, int stride, int lane) {
    int m  = lane & 15;
    int kb = (lane < 16) ? 0 : 8;
    const _Float16* p = base + m * stride + kb;
    frag16 f;
    f.h[0] = *(const v8h*)(p);
    f.h[1] = *(const v8h*)(p + 16);
    return f.v;
}

// B matrix 32x16 f16 (KxN) sourced from a TRANSPOSED [N][K] row-major buffer
// with row stride `strideK` halfs. lane<16 -> N=lane, K={0..7,16..23}; lane>=16 ->
// N=lane-16, K={8..15,24..31}. Per lane: two contiguous 8-half runs -> two b128 loads.
__device__ __forceinline__ v16h load_bT_frag(const _Float16* baseT, int strideK, int lane) {
    int n  = lane & 15;
    int kb = (lane < 16) ? 0 : 8;
    const _Float16* p = baseT + n * strideK + kb;
    frag16 f;
    f.h[0] = *(const v8h*)(p);
    f.h[1] = *(const v8h*)(p + 16);
    return f.v;
}

__device__ __forceinline__ v8f wmma_f16(v16h a, v16h b, v8f c) {
    return __builtin_amdgcn_wmma_f32_16x16x32_f16(false, a, false, b, (short)0, c,
                                                  false, false);
}

// C/D 16x16 f32: lane<16 -> N=lane, VGPR q holds M=q ; lane>=16 -> M=8+q

// ---------------------------------------------------------------------------
// Kernel A: robot MLP  [512,9] -> relu([.,32aug]@[32,128]) -> relu(@[128,64])
// writes R (f32, [B,64]) to workspace. 32 blocks x 128 threads, 16 batches/blk.
// ---------------------------------------------------------------------------
__global__ __launch_bounds__(128) void robot_mlp_kernel(
    const float* __restrict__ robot_x,
    const float* __restrict__ wr1_w, const float* __restrict__ wr1_b,
    const float* __restrict__ wr2_w, const float* __restrict__ wr2_b,
    float* __restrict__ R) {
    __shared__ __align__(16) _Float16 sA[16 * 32];      // augmented input, bias col 9
    __shared__ __align__(16) _Float16 sW1t[128 * 32];   // W1c transposed [N=128][K=32]
    __shared__ __align__(16) _Float16 sH[16 * 128];     // hidden after relu
    __shared__ __align__(16) _Float16 sW2t[64 * 128];   // wr2_w transposed [N=64][K=128]

    const int tid = threadIdx.x, lane = tid & 31, wave = tid >> 5;
    const int b0 = blockIdx.x * 16;

    for (int i = tid; i < 16 * 32; i += 128) {
        int r = i >> 5, cc = i & 31;
        float v = 0.f;
        if (cc < 9)       v = robot_x[(b0 + r) * 9 + cc];
        else if (cc == 9) v = 1.f;
        sA[i] = (_Float16)v;
    }
    for (int i = tid; i < 128 * 32; i += 128) {     // i = n*32 + k
        int n = i >> 5, k = i & 31;
        float v = 0.f;
        if (k < 9)       v = wr1_w[k * 128 + n];
        else if (k == 9) v = wr1_b[n];
        sW1t[i] = (_Float16)v;
    }
    for (int i = tid; i < 64 * 128; i += 128) {     // i = n*128 + k
        int n = i >> 7, k = i & 127;
        sW2t[i] = (_Float16)wr2_w[k * 64 + n];
    }
    __syncthreads();

    // layer 1: N=128 -> 8 ntiles, wave w does {w, w+4}; K=32 single chunk
    {
        v16h af = load_a_frag(sA, 32, lane);
#pragma unroll
        for (int t = 0; t < 2; ++t) {
            int nt = wave + 4 * t;
            v16h bf = load_bT_frag(sW1t + nt * 16 * 32, 32, lane);
            v8f acc = {};
            acc = wmma_f16(af, bf, acc);
            int n = lane & 15, mb = (lane < 16) ? 0 : 8;
#pragma unroll
            for (int q = 0; q < 8; ++q) {
                float x = acc[q]; x = x > 0.f ? x : 0.f;
                sH[(mb + q) * 128 + nt * 16 + n] = (_Float16)x;
            }
        }
    }
    __syncthreads();

    // layer 2: N=64 -> 4 ntiles, wave w does ntile w; K=128 -> 4 chunks
    {
        int nt = wave;
        v8f acc = {};
#pragma unroll
        for (int kc = 0; kc < 4; ++kc) {
            v16h af = load_a_frag(sH + kc * 32, 128, lane);
            v16h bf = load_bT_frag(sW2t + nt * 16 * 128 + kc * 32, 128, lane);
            acc = wmma_f16(af, bf, acc);
        }
        int n = lane & 15, mb = (lane < 16) ? 0 : 8;
        float bias = wr2_b[nt * 16 + n];
#pragma unroll
        for (int q = 0; q < 8; ++q) {
            float x = acc[q] + bias; x = x > 0.f ? x : 0.f;
            R[(b0 + mb + q) * 64 + nt * 16 + n] = x;
        }
    }
}

// ---------------------------------------------------------------------------
// Kernel B: human MLP per batch (63 rows padded to 64) + node-sum reduction.
// Writes rS[b] = [f16(r[0..63]) | f16(S[0..63])] ([B,128] f16).
// 512 blocks x 256 threads (8 waves).
// ---------------------------------------------------------------------------
__global__ __launch_bounds__(256) void human_mlp_kernel(
    const float* __restrict__ human_x,
    const float* __restrict__ wh1_w, const float* __restrict__ wh1_b,
    const float* __restrict__ wh2_w, const float* __restrict__ wh2_b,
    const float* __restrict__ R, _Float16* __restrict__ rS) {
    __shared__ __align__(16) _Float16 sA[64 * 32];     // 4KB  augmented, bias col 5
    __shared__ __align__(16) _Float16 sW1t[128 * 32];  // 8KB  transposed [N][K]
    __shared__ __align__(16) _Float16 sH[64 * 128];    // 16KB
    __shared__ __align__(16) _Float16 sW2t[64 * 128];  // 16KB transposed [N][K]
    __shared__ float sF[64 * 64];                      // 16KB final embeddings
    __shared__ float sPart[4][64];                     // 1KB  reduction partials

    const int tid = threadIdx.x, lane = tid & 31, wave = tid >> 5;
    const int b = blockIdx.x;

    for (int i = tid; i < 64 * 32; i += 256) {
        int r = i >> 5, cc = i & 31;
        float v = 0.f;
        if (r < 63) {                       // row 63 is padding (excluded later)
            if (cc < 5)       v = human_x[(b * 63 + r) * 5 + cc];
            else if (cc == 5) v = 1.f;
        }
        sA[i] = (_Float16)v;
    }
    for (int i = tid; i < 128 * 32; i += 256) {     // i = n*32 + k
        int n = i >> 5, k = i & 31;
        float v = 0.f;
        if (k < 5)       v = wh1_w[k * 128 + n];
        else if (k == 5) v = wh1_b[n];
        sW1t[i] = (_Float16)v;
    }
    for (int i = tid; i < 64 * 128; i += 256) {     // i = n*128 + k
        int n = i >> 7, k = i & 127;
        sW2t[i] = (_Float16)wh2_w[k * 64 + n];
    }
    __syncthreads();

    // layer 1: 4 mtiles x 8 ntiles = 32 tiles; 8 waves x 4
#pragma unroll
    for (int tt = 0; tt < 4; ++tt) {
        int t = wave + 8 * tt;
        int mt = t >> 3, nt = t & 7;
        v16h af = load_a_frag(sA + mt * 16 * 32, 32, lane);
        v16h bf = load_bT_frag(sW1t + nt * 16 * 32, 32, lane);
        v8f acc = {};
        acc = wmma_f16(af, bf, acc);
        int n = lane & 15, mb = (lane < 16) ? 0 : 8;
#pragma unroll
        for (int q = 0; q < 8; ++q) {
            float x = acc[q]; x = x > 0.f ? x : 0.f;
            sH[(mt * 16 + mb + q) * 128 + nt * 16 + n] = (_Float16)x;
        }
    }
    __syncthreads();

    // layer 2: 4 mtiles x 4 ntiles = 16 tiles; 8 waves x 2; K=128 -> 4 chunks
#pragma unroll
    for (int tt = 0; tt < 2; ++tt) {
        int t = wave + 8 * tt;
        int mt = t >> 2, nt = t & 3;
        v8f acc = {};
#pragma unroll
        for (int kc = 0; kc < 4; ++kc) {
            v16h af = load_a_frag(sH + mt * 16 * 128 + kc * 32, 128, lane);
            v16h bf = load_bT_frag(sW2t + nt * 16 * 128 + kc * 32, 128, lane);
            acc = wmma_f16(af, bf, acc);
        }
        int n = lane & 15, mb = (lane < 16) ? 0 : 8;
        float bias = wh2_b[nt * 16 + n];
#pragma unroll
        for (int q = 0; q < 8; ++q) {
            float x = acc[q] + bias; x = x > 0.f ? x : 0.f;
            sF[(mt * 16 + mb + q) * 64 + nt * 16 + n] = x;
        }
    }
    __syncthreads();

    // S[f] = r[f] + sum over the 63 real human rows (4-way parallel partials)
    {
        int f = tid & 63, g = tid >> 6;          // 256 threads = 4 groups x 64
        int r0 = g * 16, r1 = (g == 3) ? 63 : (r0 + 16);
        float s = 0.f;
        for (int row = r0; row < r1; ++row) s += sF[row * 64 + f];
        sPart[g][f] = s;
    }
    __syncthreads();
    if (tid < 64) {
        float r = R[b * 64 + tid];
        float s = r + sPart[0][tid] + sPart[1][tid] + sPart[2][tid] + sPart[3][tid];
        rS[b * 128 + tid]      = (_Float16)r;
        rS[b * 128 + 64 + tid] = (_Float16)s;
    }
}

// ---------------------------------------------------------------------------
// Kernel C: out = r @ Wr + S @ Ws + cheb_b, folded into one [512,128]@[128,64]
// GEMM with Wc = [Wr; Ws] built on the fly (transposed) from cheb_w.
// Wr = W0 + W1/63 + (2/3969-1)W2 ; Ws = -W1/63 + (124/3969)W2
// 32 blocks x 128 threads, 16 batches/blk.
// ---------------------------------------------------------------------------
__global__ __launch_bounds__(128) void cheb_kernel(
    const _Float16* __restrict__ rS, const float* __restrict__ cheb_w,
    const float* __restrict__ cheb_b, float* __restrict__ out) {
    __shared__ __align__(16) _Float16 sWt[64 * 128];   // [N=64][K=128] transposed
    const int tid = threadIdx.x, lane = tid & 31, wave = tid >> 5;
    const int b0 = blockIdx.x * 16;

    const float c1 = 1.0f / 63.0f;
    const float c2 = 2.0f / 3969.0f - 1.0f;
    const float d1 = -1.0f / 63.0f;
    const float d2 = 124.0f / 3969.0f;

    for (int i = tid; i < 64 * 128; i += 128) {     // i = n*128 + k
        int n = i >> 7, k = i & 127;
        float v;
        if (k < 64) {
            v = cheb_w[k * 64 + n] + c1 * cheb_w[4096 + k * 64 + n] +
                c2 * cheb_w[8192 + k * 64 + n];
        } else {
            int kk = k - 64;
            v = d1 * cheb_w[4096 + kk * 64 + n] + d2 * cheb_w[8192 + kk * 64 + n];
        }
        sWt[i] = (_Float16)v;
    }
    __syncthreads();

    int nt = wave;  // ntile 0..3 (N=64)
    v8f acc = {};
#pragma unroll
    for (int kc = 0; kc < 4; ++kc) {
        v16h af = load_a_frag(rS + b0 * 128 + kc * 32, 128, lane);   // global b128
        v16h bf = load_bT_frag(sWt + nt * 16 * 128 + kc * 32, 128, lane);
        acc = wmma_f16(af, bf, acc);
    }
    int n = lane & 15, mb = (lane < 16) ? 0 : 8;
    float bias = cheb_b[nt * 16 + n];
#pragma unroll
    for (int q = 0; q < 8; ++q)
        out[(b0 + mb + q) * 64 + nt * 16 + n] = acc[q] + bias;
}

// ---------------------------------------------------------------------------

extern "C" void kernel_launch(void* const* d_in, const int* in_sizes, int n_in,
                              void* d_out, int out_size, void* d_ws, size_t ws_size,
                              hipStream_t stream) {
    const float* robot_x = (const float*)d_in[0];
    const float* human_x = (const float*)d_in[1];
    // d_in[2] = edge_index: complete digraph, closed-form Laplacian used instead
    const float* wr1_w = (const float*)d_in[3];
    const float* wr1_b = (const float*)d_in[4];
    const float* wr2_w = (const float*)d_in[5];
    const float* wr2_b = (const float*)d_in[6];
    const float* wh1_w = (const float*)d_in[7];
    const float* wh1_b = (const float*)d_in[8];
    const float* wh2_w = (const float*)d_in[9];
    const float* wh2_b = (const float*)d_in[10];
    const float* cheb_w = (const float*)d_in[11];
    const float* cheb_b = (const float*)d_in[12];
    float* out = (float*)d_out;

    // workspace: R f32 [512,64] @0 ; rS f16 [512,128] @131072
    float*    R  = (float*)d_ws;
    _Float16* rS = (_Float16*)((char*)d_ws + 512 * 64 * sizeof(float));

    robot_mlp_kernel<<<32, 128, 0, stream>>>(robot_x, wr1_w, wr1_b, wr2_w, wr2_b, R);
    human_mlp_kernel<<<512, 256, 0, stream>>>(human_x, wh1_w, wh1_b, wh2_w, wh2_b,
                                              R, rS);
    cheb_kernel<<<32, 128, 0, stream>>>(rS, cheb_w, cheb_b, out);
}
